// cosPcTransformerMH_56264071577747
// MI455X (gfx1250) — matrix operand order repaired and checked by reading, not compiled, not run on hardware
//
#include <hip/hip_runtime.h>
#include <hip/hip_bf16.h>

// Problem constants (match reference)
#define BB 2048
#define CC 8
#define DD 2048
#define EPSV 1e-12f

typedef __bf16 bf16_t;
typedef bf16_t v16bf __attribute__((ext_vector_type(16)));
typedef bf16_t v8bf  __attribute__((ext_vector_type(8)));
typedef float  v8f   __attribute__((ext_vector_type(8)));

// ---------------------------------------------------------------------------
// fp32 -> bf16 conversion (grid-stride)
// ---------------------------------------------------------------------------
__global__ void cvt_f32_bf16(const float* __restrict__ in, bf16_t* __restrict__ out, long n) {
    long i = (long)blockIdx.x * blockDim.x + threadIdx.x;
    long stride = (long)gridDim.x * blockDim.x;
    for (; i < n; i += stride) out[i] = (bf16_t)in[i];
}

// ---------------------------------------------------------------------------
// Tiled bf16 WMMA GEMM:  C[m,n] (+= bias[n]) = sum_k A(m,k) * B(k,n)
//   TRANS_A = 0 : A(m,k) = A[m*lda + k]
//   TRANS_A = 1 : A(m,k) = A[k*lda + m]
//   B(k,n) = B[k*ldb + n]
// Per-channel (blockIdx.z) base offsets aChan/bChan/cChan (elements).
// Block tile 128x128, BK=64; 8 waves (2x4); each wave computes 64x32
// (4x2 sub-tiles of 16x16) -> 8 v_wmma per K-step with 4+2 fragment loads.
// ---------------------------------------------------------------------------
template <bool TRANS_A, typename OutT>
__global__ __launch_bounds__(256)
void gemm_wmma_bf16(const bf16_t* __restrict__ Ab, long aChan, long lda,
                    const bf16_t* __restrict__ Bb, long bChan, long ldb,
                    OutT* __restrict__ Cb, long cChan, long ldc,
                    const float* __restrict__ bias, int K) {
    __shared__ __attribute__((aligned(32))) bf16_t As[128 * 64];
    __shared__ __attribute__((aligned(32))) bf16_t Bs[128 * 64];

    const int ch = blockIdx.z;
    const bf16_t* A = Ab + (long)ch * aChan;
    const bf16_t* B = Bb + (long)ch * bChan;
    OutT* C = Cb + (long)ch * cChan;

    const int m0 = blockIdx.y * 128;
    const int n0 = blockIdx.x * 128;
    const int tid = threadIdx.x;
    const int lane = tid & 31;
    const int wid = tid >> 5;  // 0..7
    const int wm = wid >> 2;   // 0..1  -> 64-row slab
    const int wn = wid & 3;    // 0..3  -> 32-col slab

    v8f acc[4][2];
#pragma unroll
    for (int i = 0; i < 4; ++i)
#pragma unroll
        for (int j = 0; j < 2; ++j)
            acc[i][j] = (v8f){0.f, 0.f, 0.f, 0.f, 0.f, 0.f, 0.f, 0.f};

    // per-lane fragment coordinates (ISA 7.12.2 layouts, wave32)
    const int mA = lane & 15;                 // A: row within 16x32 tile
    const int khalfA = (lane >> 4) * 8;       // A: K group 0..7 / 8..15 (then +16)
    const int nBcol = lane & 15;              // B: column within KxN tile
    const int kbaseB = (lane >> 4) * 8;       // B: K group 0..7 / 8..15 (then +16)

    for (int k0 = 0; k0 < K; k0 += 64) {
        __syncthreads();
        // ---- stage A tile (128 x 64) into LDS ----
        if (!TRANS_A) {
#pragma unroll
            for (int i = 0; i < 4; ++i) {
                int idx = tid + i * 256;          // 0..1023
                int r = idx >> 3;                 // 0..127
                int kc = (idx & 7) * 8;           // 0..56
                v8bf v = *(const v8bf*)(A + (long)(m0 + r) * lda + (k0 + kc));
                *(v8bf*)(As + r * 64 + kc) = v;
            }
        } else {
#pragma unroll
            for (int i = 0; i < 4; ++i) {
                int idx = tid + i * 256;          // 0..1023
                int kk = idx >> 4;                // 0..63
                int mv = (idx & 15) * 8;          // 0..120
                v8bf v = *(const v8bf*)(A + (long)(k0 + kk) * lda + (m0 + mv));
#pragma unroll
                for (int e = 0; e < 8; ++e) As[(mv + e) * 64 + kk] = v[e];
            }
        }
        // ---- stage B tile (64k x 128n) transposed into LDS: Bs[n][k] ----
#pragma unroll
        for (int i = 0; i < 4; ++i) {
            int idx = tid + i * 256;              // 0..1023
            int kk = idx >> 4;                    // 0..63
            int nv = (idx & 15) * 8;              // 0..120
            v8bf v = *(const v8bf*)(B + (long)(k0 + kk) * ldb + (n0 + nv));
#pragma unroll
            for (int e = 0; e < 8; ++e) Bs[(nv + e) * 64 + kk] = v[e];
        }

        // ---- prefetch next K-tile into cache (global_prefetch_b8) ----
        if (k0 + 64 < K) {
            if (!TRANS_A) {
                int r = tid >> 1, kc = (tid & 1) * 32;
                __builtin_prefetch(A + (long)(m0 + r) * lda + (k0 + 64 + kc), 0, 0);
            } else {
                int kk = tid >> 2, mv = (tid & 3) * 32;
                __builtin_prefetch(A + (long)(k0 + 64 + kk) * lda + (m0 + mv), 0, 0);
            }
            int kk = tid >> 2, nv = (tid & 3) * 32;
            __builtin_prefetch(B + (long)(k0 + 64 + kk) * ldb + (n0 + nv), 0, 0);
        }
        __syncthreads();

        // ---- two 16x16x32 WMMA K-steps over the 64-deep LDS tile ----
#pragma unroll
        for (int ks = 0; ks < 64; ks += 32) {
            v16bf afr[4], bfr[2];
#pragma unroll
            for (int mi = 0; mi < 4; ++mi) {
                int m = wm * 64 + mi * 16 + mA;
                v8bf lo = *(const v8bf*)(As + m * 64 + ks + khalfA);
                v8bf hi = *(const v8bf*)(As + m * 64 + ks + khalfA + 16);
                afr[mi] = __builtin_shufflevector(lo, hi, 0, 1, 2, 3, 4, 5, 6, 7,
                                                  8, 9, 10, 11, 12, 13, 14, 15);
            }
#pragma unroll
            for (int ni = 0; ni < 2; ++ni) {
                int n = wn * 32 + ni * 16 + nBcol;
                v8bf blo = *(const v8bf*)(Bs + n * 64 + ks + kbaseB);
                v8bf bhi = *(const v8bf*)(Bs + n * 64 + ks + kbaseB + 16);
                bfr[ni] = __builtin_shufflevector(blo, bhi, 0, 1, 2, 3, 4, 5, 6, 7,
                                                  8, 9, 10, 11, 12, 13, 14, 15);
            }
#pragma unroll
            for (int mi = 0; mi < 4; ++mi)
#pragma unroll
                for (int ni = 0; ni < 2; ++ni)
                    acc[mi][ni] = __builtin_amdgcn_wmma_f32_16x16x32_bf16(
                        false, afr[mi], false, bfr[ni], (short)0, acc[mi][ni],
                        false, false);
        }
    }

    // ---- epilogue: C/D layout — element r -> row r (+8 for lanes 16..31) ----
    const int nOut = lane & 15;
    const int mOff = (lane >> 4) * 8;
#pragma unroll
    for (int mi = 0; mi < 4; ++mi)
#pragma unroll
        for (int ni = 0; ni < 2; ++ni) {
            int mbase = m0 + wm * 64 + mi * 16 + mOff;
            int nn = n0 + wn * 32 + ni * 16 + nOut;
            float bval = bias ? bias[(long)ch * DD + nn] : 0.f;
#pragma unroll
            for (int r = 0; r < 8; ++r) {
                float v = acc[mi][ni][r] + bval;
                C[(long)(mbase + r) * ldc + nn] = (OutT)v;
            }
        }
}

// ---------------------------------------------------------------------------
// Batch-axis squared L2 norms: DQ2[c*D+o] = sum_b Q[b,c,o]^2 (likewise K)
// ---------------------------------------------------------------------------
__global__ void norms_kernel(const bf16_t* __restrict__ Qb, const bf16_t* __restrict__ Kb,
                             float* __restrict__ DQ2, float* __restrict__ DK2) {
    const long CD = (long)CC * DD;
    int t = blockIdx.x * blockDim.x + threadIdx.x;   // 0 .. 2*C*D-1
    int which = t / (CC * DD);
    int co = t % (CC * DD);
    const bf16_t* P = which ? Kb : Qb;
    float s = 0.f;
    for (int b = 0; b < BB; ++b) {
        float v = (float)P[(long)b * CD + co];
        s += v * v;
    }
    (which ? DK2 : DQ2)[co] = s;
}

// ---------------------------------------------------------------------------
// Cosine-scale + online softmax over first axis of S[o][c][p]; emit bf16 SM.
// One thread per (c, p) column; strided walk over o. Two passes over S.
// ---------------------------------------------------------------------------
__global__ void scale_softmax_kernel(const float* __restrict__ S,
                                     const float* __restrict__ DQ2,
                                     const float* __restrict__ DK2,
                                     bf16_t* __restrict__ SMb) {
    const long CD = (long)CC * DD;
    int t = blockIdx.x * blockDim.x + threadIdx.x;   // 0 .. C*D-1
    int c = t / DD;
    int j = t % DD;
    const float dk2 = DK2[(long)c * DD + j];
    const long base = (long)c * DD + j;

    // pass 1: online max + rescaled sum
    float mx = -3.402823466e38f;
    float sum = 0.f;
    for (int i = 0; i < DD; ++i) {
        float sc = fmaxf(sqrtf(DQ2[(long)c * DD + i] * dk2), EPSV);
        float y = S[(long)i * CD + base] / sc;
        float nm = fmaxf(mx, y);
        sum = sum * expf(mx - nm) + expf(y - nm);
        mx = nm;
    }
    float inv = 1.f / sum;
    // pass 2: write normalized bf16
    for (int i = 0; i < DD; ++i) {
        float sc = fmaxf(sqrtf(DQ2[(long)c * DD + i] * dk2), EPSV);
        float y = S[(long)i * CD + base] / sc;
        SMb[(long)i * CD + base] = (bf16_t)(expf(y - mx) * inv);
    }
}

// ---------------------------------------------------------------------------
// Host-side orchestration
// ---------------------------------------------------------------------------
extern "C" void kernel_launch(void* const* d_in, const int* in_sizes, int n_in,
                              void* d_out, int out_size, void* d_ws, size_t ws_size,
                              hipStream_t stream) {
    const float* x   = (const float*)d_in[0];
    const float* Wq  = (const float*)d_in[1];
    const float* Wq0 = (const float*)d_in[2];
    const float* Wk  = (const float*)d_in[3];
    const float* Wk0 = (const float*)d_in[4];
    float* out = (float*)d_out;

    const long CD  = (long)CC * DD;          // 16384
    const long NBC = (long)BB * CD;          // 33,554,432 (= C*D*D too)
    const size_t SZ_BF = (size_t)NBC * sizeof(bf16_t);   // 64 MiB

    char* ws = (char*)d_ws;
    bf16_t* xb  = (bf16_t*)(ws + 0 * SZ_BF);
    bf16_t* wqb = (bf16_t*)(ws + 1 * SZ_BF);
    bf16_t* wkb = (bf16_t*)(ws + 2 * SZ_BF);
    bf16_t* Qb  = (bf16_t*)(ws + 3 * SZ_BF);
    bf16_t* Kb  = (bf16_t*)(ws + 4 * SZ_BF);
    // scores (fp32) reuse the dead weight space (needs 2*SZ_BF: exactly fits)
    float*  S   = (float*)(ws + 1 * SZ_BF);
    // softmax output (bf16) reuses dead Qb space
    bf16_t* SMb = (bf16_t*)(ws + 3 * SZ_BF);
    float* DQ2 = (float*)(ws + 5 * SZ_BF);
    float* DK2 = DQ2 + CD;

    // 1) convert inputs to bf16
    cvt_f32_bf16<<<4096, 256, 0, stream>>>(x,  xb,  NBC);
    cvt_f32_bf16<<<4096, 256, 0, stream>>>(Wq, wqb, NBC);
    cvt_f32_bf16<<<4096, 256, 0, stream>>>(Wk, wkb, NBC);

    dim3 ggrid(DD / 128, BB / 128, CC);  // (16, 16, 8)
    dim3 gblk(256);

    // 2) projections: Q = X*Wq + Wq0 ; K = X*Wk + Wk0   (bf16 out)
    gemm_wmma_bf16<false, bf16_t><<<ggrid, gblk, 0, stream>>>(
        xb, DD, CD, wqb, (long)DD * DD, DD, Qb, DD, CD, Wq0, DD);
    gemm_wmma_bf16<false, bf16_t><<<ggrid, gblk, 0, stream>>>(
        xb, DD, CD, wkb, (long)DD * DD, DD, Kb, DD, CD, Wk0, DD);

    // 3) batch-axis norms
    norms_kernel<<<(2 * CC * DD) / 256, 256, 0, stream>>>(Qb, Kb, DQ2, DK2);

    // 4) scores: S[o,c,p] = sum_b Q[b,c,o]*K[b,c,p]   (A transposed, fp32 out)
    gemm_wmma_bf16<true, float><<<ggrid, gblk, 0, stream>>>(
        Qb, DD, CD, Kb, DD, CD, S, DD, CD, nullptr, BB);

    // 5) cosine scale + online softmax over first axis -> bf16 SM
    scale_softmax_kernel<<<(CC * DD) / 256, 256, 0, stream>>>(S, DQ2, DK2, SMb);

    // 6) Z[b,c,o] = sum_p x[b,c,p] * SM[p,c,o]   (fp32 out to d_out)
    gemm_wmma_bf16<false, float><<<ggrid, gblk, 0, stream>>>(
        xb, DD, CD, SMb, DD, CD, out, DD, CD, nullptr, DD);
}
